// AEEncoder_66340064854757
// MI455X (gfx1250) — matrix operand order repaired and checked by reading, not compiled, not run on hardware
//
#include <hip/hip_runtime.h>

typedef __attribute__((ext_vector_type(16))) _Float16 v16h;
typedef __attribute__((ext_vector_type(4)))  _Float16 v4h;
typedef __attribute__((ext_vector_type(8)))  float    v8f;
typedef __attribute__((ext_vector_type(4)))  float    v4f;

#define BATCH  256
#define NGENES 8192
#define WM     4
#define HID    32768
#define NTF    1024
#define GPT    64
#define EPSBN  1e-5f

// ---------------------------------------------------------------------------
// Kernel A: fused  L1 -> ReLU -> BN -> L2 -> ReLU -> BN  , output f16 h2.
// One workgroup (256 thr) owns 32 hidden features x all 256 batch rows, so
// both BatchNorms reduce entirely in LDS.
// ---------------------------------------------------------------------------
__global__ __launch_bounds__(256) void fused_l12(
    const float* __restrict__ x,  const float* __restrict__ w1,
    const float* __restrict__ b1, const float* __restrict__ w2,
    const float* __restrict__ b2, _Float16* __restrict__ h2)
{
    __shared__ float A1[32][257];          // relu(L1) staged, padded
    __shared__ float P[8][32], PSS[8][32]; // partial sums per (bsub, f)
    __shared__ float M1[32], R1[32];       // BN1 mean / rsqrt(var)
    __shared__ float M2[32], R2[32];       // BN2 mean / rsqrt(var)

    const int tid  = threadIdx.x;
    const int f    = tid & 31;
    const int bsub = tid >> 5;
    const int i    = blockIdx.x * 32 + f;
    const int gene = i >> 2;

    const float wi = w1[i], bi = b1[i];

    float s = 0.f, ss = 0.f;
    #pragma unroll 4
    for (int k = 0; k < 32; ++k) {
        const int b = bsub * 32 + k;
        float v = fmaf(wi, x[(size_t)b * NGENES + gene], bi);
        v = fmaxf(v, 0.f);
        A1[f][b] = v;
        s += v; ss += v * v;
    }
    P[bsub][f] = s; PSS[bsub][f] = ss;
    __syncthreads();
    if (tid < 32) {
        float t = 0.f, tss = 0.f;
        #pragma unroll
        for (int k = 0; k < 8; ++k) { t += P[k][tid]; tss += PSS[k][tid]; }
        const float m = t * (1.f / 256.f);
        M1[tid] = m;
        R1[tid] = rsqrtf(tss * (1.f / 256.f) - m * m + EPSBN);
    }
    __syncthreads();

    const int   gb  = f & ~3;
    const float w20 = w2[i*4+0], w21 = w2[i*4+1], w22 = w2[i*4+2], w23 = w2[i*4+3];
    const float m0 = M1[gb+0], r0 = R1[gb+0];
    const float m1 = M1[gb+1], r1 = R1[gb+1];
    const float m2 = M1[gb+2], r2 = R1[gb+2];
    const float m3 = M1[gb+3], r3 = R1[gb+3];
    const float b2i = b2[i];

    s = 0.f; ss = 0.f;
    #pragma unroll 4
    for (int k = 0; k < 32; ++k) {
        const int b = bsub * 32 + k;
        float v = b2i;
        v = fmaf(w20, (A1[gb+0][b] - m0) * r0, v);
        v = fmaf(w21, (A1[gb+1][b] - m1) * r1, v);
        v = fmaf(w22, (A1[gb+2][b] - m2) * r2, v);
        v = fmaf(w23, (A1[gb+3][b] - m3) * r3, v);
        v = fmaxf(v, 0.f);
        s += v; ss += v * v;
    }
    P[bsub][f] = s; PSS[bsub][f] = ss;
    __syncthreads();
    if (tid < 32) {
        float t = 0.f, tss = 0.f;
        #pragma unroll
        for (int k = 0; k < 8; ++k) { t += P[k][tid]; tss += PSS[k][tid]; }
        const float m = t * (1.f / 256.f);
        M2[tid] = m;
        R2[tid] = rsqrtf(tss * (1.f / 256.f) - m * m + EPSBN);
    }
    __syncthreads();

    const float mm = M2[f], rr = R2[f];
    #pragma unroll 4
    for (int k = 0; k < 32; ++k) {
        const int b = bsub * 32 + k;
        float v = b2i;
        v = fmaf(w20, (A1[gb+0][b] - m0) * r0, v);
        v = fmaf(w21, (A1[gb+1][b] - m1) * r1, v);
        v = fmaf(w22, (A1[gb+2][b] - m2) * r2, v);
        v = fmaf(w23, (A1[gb+3][b] - m3) * r3, v);
        v = fmaxf(v, 0.f);
        h2[(size_t)b * HID + i] = (_Float16)((v - mm) * rr);
    }
}

// ---------------------------------------------------------------------------
// Kernel B: Layer 3 as gathered block-dense WMMA GEMM + fused BatchNorm,
// software-pipelined with double-buffered LDS.
//   * cols3 structure => contiguous gene blocks of WM=4 f16: A chunk
//     [256 x 64] gathered with b64 loads/stores (16/thread/chunk).
//   * B tile has exactly one live row (transposed storage, row == tfl):
//     zeroed once, only the 64 live halves rewritten per chunk (16 threads
//     x one packed v4h store), stale row re-zeroed when tfl advances.
//   * Each iteration issues the NEXT chunk's gather before the current
//     chunk's WMMAs, one barrier per chunk: VMEM gather latency hides under
//     ds_load_b128 + v_wmma_f32_16x16x32_f16.
// b3 omitted: per-column shift cancels under affine-free BatchNorm.
// ---------------------------------------------------------------------------
__device__ __forceinline__ void gather_chunk(
    int tid, int base, int tfl, int prev_tfl,
    const _Float16* __restrict__ h2, const int* __restrict__ cols3,
    const float* __restrict__ w3,
    _Float16 (*__restrict__ A)[72], _Float16 (*__restrict__ B)[72])
{
    // A: thread == batch row; 16 gene-blocks of 4 f16 each (8B, aligned)
    const _Float16* hrow = h2 + (size_t)tid * HID;
    unsigned long long* arow = (unsigned long long*)&A[tid][0];
    #pragma unroll
    for (int g = 0; g < 16; ++g) {
        const int c = cols3[base + g * 4];          // uniform -> scalar load
        arow[g] = *(const unsigned long long*)(hrow + c);
    }
    // B: only row 'tfl' is live; 16 threads write 4 halves each
    if (tid < 16) {
        const v4f w = *(const v4f*)(w3 + base + tid * 4);
        v4h hv;
        hv[0] = (_Float16)w[0]; hv[1] = (_Float16)w[1];
        hv[2] = (_Float16)w[2]; hv[3] = (_Float16)w[3];
        *(v4h*)&B[tfl][tid * 4] = hv;
        if (prev_tfl >= 0 && prev_tfl != tfl)
            *(unsigned long long*)&B[prev_tfl][tid * 4] = 0ull;
    }
}

__global__ __launch_bounds__(256) void l3_wmma(
    const _Float16* __restrict__ h2, const float* __restrict__ w3,
    const int* __restrict__ cols3, float* __restrict__ out)
{
    __shared__ _Float16 Asm[2][256][72];   // double-buffered A chunks (72 KB)
    __shared__ _Float16 Bsm[2][16][72];    // double-buffered B tiles
    __shared__ float    CS[16], CSS[16];

    const int tid   = threadIdx.x;
    const int lane  = tid & 31;
    const int wv    = tid >> 5;
    const int n     = lane & 15;           // output column within tile
    const int hi    = lane >> 4;           // lane half selects K sub-range
    const int tbase = blockIdx.x * 16;     // first TF of this group

    const int rowA0 = (wv * 2 + 0) * 16 + (lane & 15);
    const int rowA1 = (wv * 2 + 1) * 16 + (lane & 15);

    if (tid < 16) { CS[tid] = 0.f; CSS[tid] = 0.f; }
    {   // zero both B buffers ([16][32] region = 128 x 8B per buffer)
        const int pb = tid >> 7, idx = tid & 127;
        ((unsigned long long*)&Bsm[pb][idx >> 3][0])[idx & 7] = 0ull;
    }
    __syncthreads();

    // prologue: stage chunk 0 into buffer 0
    gather_chunk(tid, tbase * 256, 0, -1, h2, cols3, w3, Asm[0], Bsm[0]);
    __syncthreads();

    v8f c0 = {}, c1 = {};
    int p = 0;

    for (int kc = 0; kc < 16 * 256; kc += 64) {
        // ---- issue next chunk's gather into the other buffer ----
        const int kcn = kc + 64;
        if (kcn < 16 * 256) {
            const int tfln  = kcn >> 8;
            const int basen = (tbase + tfln) * 256 + (kcn & 255);
            const int ptfl  = (kcn >= 128) ? ((kcn - 128) >> 8) : -1;
            __builtin_prefetch(cols3 + basen + 64, 0, 1);
            gather_chunk(tid, basen, tfln, ptfl, h2, cols3, w3,
                         Asm[1 - p], Bsm[1 - p]);
        }

        // ---- WMMA on current buffer ----
        #pragma unroll
        for (int ks = 0; ks < 2; ++ks) {
            const _Float16* ap0 = &Asm[p][rowA0][ks * 32 + hi * 8];
            const _Float16* ap1 = &Asm[p][rowA1][ks * 32 + hi * 8];
            const _Float16* bp  = &Bsm[p][n][ks * 32 + hi * 8];
            v16h a0, a1, b;
            #pragma unroll
            for (int u = 0; u < 8; ++u) {
                a0[u]     = ap0[u];      // K 0-7   (or 8-15)
                a0[u + 8] = ap0[16 + u]; // K 16-23 (or 24-31)
                a1[u]     = ap1[u];
                a1[u + 8] = ap1[16 + u];
                b[u]      = bp[u];       // same K split as A
                b[u + 8]  = bp[16 + u];
            }
            c0 = __builtin_amdgcn_wmma_f32_16x16x32_f16(
                     false, a0, false, b, (short)0, c0, false, false);
            c1 = __builtin_amdgcn_wmma_f32_16x16x32_f16(
                     false, a1, false, b, (short)0, c1, false, false);
        }

        __syncthreads();                   // one barrier per chunk
        p ^= 1;
    }

    // ---- fused BatchNorm over batch (whole batch lives in this WG) ----
    float s = 0.f, ss = 0.f;
    #pragma unroll
    for (int r = 0; r < 8; ++r) {
        s  += c0[r] + c1[r];
        ss += c0[r] * c0[r] + c1[r] * c1[r];
    }
    atomicAdd(&CS[n],  s);
    atomicAdd(&CSS[n], ss);
    __syncthreads();

    const float mean = CS[n] * (1.f / 256.f);
    const float rs   = rsqrtf(CSS[n] * (1.f / 256.f) - mean * mean + EPSBN);

    #pragma unroll
    for (int r = 0; r < 8; ++r) {
        const int b0 = (wv * 2 + 0) * 16 + r + hi * 8;  // C layout: M = r+8*hi
        const int b1 = (wv * 2 + 1) * 16 + r + hi * 8;
        out[(size_t)b0 * NTF + tbase + n] = (c0[r] - mean) * rs;
        out[(size_t)b1 * NTF + tbase + n] = (c1[r] - mean) * rs;
    }
}

// ---------------------------------------------------------------------------
extern "C" void kernel_launch(void* const* d_in, const int* in_sizes, int n_in,
                              void* d_out, int out_size, void* d_ws, size_t ws_size,
                              hipStream_t stream)
{
    const float* x   = (const float*)d_in[0];
    const float* w1  = (const float*)d_in[1];
    const float* b1  = (const float*)d_in[2];
    const float* w2  = (const float*)d_in[3];
    const float* b2  = (const float*)d_in[4];
    const float* w3  = (const float*)d_in[5];
    // d_in[6] = b3: per-column shift, cancels under affine-free BatchNorm.
    const int* cols3 = (const int*)d_in[12];

    _Float16* h2 = (_Float16*)d_ws;          // 256 x 32768 f16 = 16 MB
    float*    out = (float*)d_out;           // 256 x 1024 f32

    fused_l12<<<HID / 32, 256, 0, stream>>>(x, w1, b1, w2, b2, h2);
    l3_wmma  <<<NTF / 16, 256, 0, stream>>>(h2, w3, cols3, out);
}
